// Pool_7413113552901
// MI455X (gfx1250) — matrix operand order repaired and checked by reading, not compiled, not run on hardware
//
#include <hip/hip_runtime.h>
#include <math.h>

// ---------------------------------------------------------------------------
// GNN pipeline: 3x (GENConv softmax-aggr + TopKPool) + mean-pool + MLP head.
// Dense linear algebra: LDS-tiled v_wmma_f32_16x16x32_f16 GEMM (f16 operands,
// f32 accumulate), 64x64 block tile, 2 WMMA per wave per K-step, fragments
// loaded from LDS as b128. Edge softmax aggregation is bandwidth-bound ->
// fused recompute + atomics, no msg[E,C] materialization.
// ---------------------------------------------------------------------------

typedef __attribute__((ext_vector_type(16))) _Float16 v16h;
typedef __attribute__((ext_vector_type(8)))  _Float16 v8h;
typedef __attribute__((ext_vector_type(8)))  float    v8f;

#define GG   128
#define NPG  1024
#define EE   1048576
#define NN   (GG*NPG)     // 131072
#define KK1  205
#define KK2  41
#define KK3  9
#define N1   (GG*KK1)     // 26240
#define N2   (GG*KK2)     // 5248
#define N3   (GG*KK3)     // 1152

#define BM 64
#define BN 64
#define BK 32

static inline int cdiv(long long a, int b) { return (int)((a + b - 1) / b); }

// ------------------------------ fill kernels -------------------------------
__global__ void fill_f32_kernel(float* p, float v, int n) {
  int i = blockIdx.x * blockDim.x + threadIdx.x;
  if (i < n) p[i] = v;
}
__global__ void fill_i32_kernel(int* p, int v, int n) {
  int i = blockIdx.x * blockDim.x + threadIdx.x;
  if (i < n) p[i] = v;
}

// ------------------------------ WMMA GEMM ----------------------------------
// C[M,Ncols] = A[M,K] @ B[K,Ncols] + bias[Ncols]
// Block (256 thr / 8 waves) -> 64x64 tile. Wave w: rows (w>>1)*16, cols
// (w&1)*32 (two 16x16 WMMA subtiles). A staged row-major f16, B staged
// transposed f16 so both fragments are contiguous 16B chunks (ds_load_b128):
//   A frag lane: K = 8*half+0..7 at rowptr+8*half, K = 16+8*half+0..7 at +16
//   B frag lane: K = 8*half+0..7 at colptr+8*half, K = 16+8*half+0..7 at +16
// (fragment layouts per CDNA5 ISA 7.12.2)
__global__ void __launch_bounds__(256)
wmma_gemm_kernel(const float* __restrict__ A, const float* __restrict__ B,
                 const float* __restrict__ bias, float* __restrict__ C,
                 int M, int Ncols, int K) {
  __shared__ __attribute__((aligned(16))) _Float16 la[BM * BK];   // [64][32]
  __shared__ __attribute__((aligned(16))) _Float16 lbt[BN * BK];  // [64 cols][32 k]

  int tilesN = Ncols / BN;
  int bm = blockIdx.x / tilesN;
  int bn = blockIdx.x - bm * tilesN;
  int m0 = bm * BM, n0 = bn * BN;

  int tid  = threadIdx.x;
  int wv   = tid >> 5;
  int lane = tid & 31;
  int wm   = wv >> 1;        // 0..3: 16-row subtile
  int wn   = wv & 1;         // 0..1: 32-col strip
  int half = lane >> 4;
  int l15  = lane & 15;

  v8f acc0 = {0.f,0.f,0.f,0.f,0.f,0.f,0.f,0.f};
  v8f acc1 = {0.f,0.f,0.f,0.f,0.f,0.f,0.f,0.f};

  for (int k0 = 0; k0 < K; k0 += BK) {
    // prefetch next A K-chunk (lowers to global_prefetch_b8)
    if (k0 + BK < K)
      __builtin_prefetch(&A[(long long)(m0 + (tid >> 2)) * K + k0 + BK], 0, 1);

    // stage A tile (64x32) -> f16, coalesced 32-wide rows
#pragma unroll
    for (int i = 0; i < (BM * BK) / 256; ++i) {
      int li = tid + i * 256;
      int r = li >> 5, c = li & 31;
      la[li] = (_Float16)A[(long long)(m0 + r) * K + k0 + c];
    }
    // stage B tile (32x64) transposed -> f16, coalesced 64-wide rows
#pragma unroll
    for (int i = 0; i < (BK * BN) / 256; ++i) {
      int li = tid + i * 256;
      int kr = li >> 6, nc = li & 63;
      lbt[nc * BK + kr] = (_Float16)B[(long long)(k0 + kr) * Ncols + n0 + nc];
    }
    __syncthreads();

    // A fragment: two contiguous 16B chunks
    const _Float16* ap = &la[(wm * 16 + l15) * BK + 8 * half];
    v8h alo = *(const v8h*)(ap);
    v8h ahi = *(const v8h*)(ap + 16);
    v16h af = __builtin_shufflevector(alo, ahi, 0,1,2,3,4,5,6,7,8,9,10,11,12,13,14,15);

    // B fragments for the two 16-col subtiles: two 16B chunks per lane
    const _Float16* bp0 = &lbt[(wn * 32 + l15) * BK + 8 * half];
    const _Float16* bp1 = &lbt[(wn * 32 + 16 + l15) * BK + 8 * half];
    v8h b0l = *(const v8h*)(bp0);
    v8h b0h = *(const v8h*)(bp0 + 16);
    v8h b1l = *(const v8h*)(bp1);
    v8h b1h = *(const v8h*)(bp1 + 16);
    v16h bf0 = __builtin_shufflevector(b0l, b0h, 0,1,2,3,4,5,6,7,8,9,10,11,12,13,14,15);
    v16h bf1 = __builtin_shufflevector(b1l, b1h, 0,1,2,3,4,5,6,7,8,9,10,11,12,13,14,15);

    acc0 = __builtin_amdgcn_wmma_f32_16x16x32_f16(false, af, false, bf0,
                                                  (short)0, acc0, false, false);
    acc1 = __builtin_amdgcn_wmma_f32_16x16x32_f16(false, af, false, bf1,
                                                  (short)0, acc1, false, false);
    __syncthreads();
  }

  // C/D layout: N = lane&15, M = r + 8*half
  int ccol = n0 + wn * 32 + l15;
  float bia0 = bias[ccol];
  float bia1 = bias[ccol + 16];
#pragma unroll
  for (int r = 0; r < 8; ++r) {
    int row = m0 + wm * 16 + r + 8 * half;
    C[(long long)row * Ncols + ccol]      = acc0[r] + bia0;
    C[(long long)row * Ncols + ccol + 16] = acc1[r] + bia1;
  }
}

static void launch_gemm(hipStream_t s, const float* A, const float* B,
                        const float* bias, float* C, int M, int Ncols, int K) {
  int blocks = (M / BM) * (Ncols / BN);
  wmma_gemm_kernel<<<blocks, 256, 0, s>>>(A, B, bias, C, M, Ncols, K);
}

// --------------------------- edge message passes ---------------------------
// msg(e,c) = relu(xs[src[e]][c] + dot(ea[e], we[:,c]) + be[c]) + 1e-7  (> 0)
// Phase 1: per-dst channel max via integer atomicMax (valid: msg > 0, init 0;
//          zero-init also realizes the reference's isfinite->0 fallback).
__global__ void edge_max_kernel(const float* __restrict__ xs,
                                const float* __restrict__ ea,
                                const float* __restrict__ we,
                                const float* __restrict__ be,
                                const int* __restrict__ src,
                                const int* __restrict__ dst,
                                const int* __restrict__ ev,
                                float* __restrict__ mx, int ne, int C) {
  int t = blockIdx.x * blockDim.x + threadIdx.x;
  if (t >= ne * C) return;
  int e = t / C, c = t - e * C;
  if (!ev[e]) return;
  float s = be[c];
  const float* ar = ea + (long long)e * 32;
#pragma unroll 8
  for (int k = 0; k < 32; ++k) s = fmaf(ar[k], we[k * C + c], s);
  float m = fmaxf(xs[(long long)src[e] * C + c] + s, 0.f) + 1e-7f;
  atomicMax((int*)(mx + (long long)dst[e] * C + c), __float_as_int(m));
}

// Phase 2: recompute msg, accumulate exp sums (denominator + weighted sum).
__global__ void edge_sum_kernel(const float* __restrict__ xs,
                                const float* __restrict__ ea,
                                const float* __restrict__ we,
                                const float* __restrict__ be,
                                const int* __restrict__ src,
                                const int* __restrict__ dst,
                                const int* __restrict__ ev,
                                const float* __restrict__ mx,
                                float* __restrict__ den,
                                float* __restrict__ num, int ne, int C) {
  int t = blockIdx.x * blockDim.x + threadIdx.x;
  if (t >= ne * C) return;
  int e = t / C, c = t - e * C;
  if (!ev[e]) return;
  float s = be[c];
  const float* ar = ea + (long long)e * 32;
#pragma unroll 8
  for (int k = 0; k < 32; ++k) s = fmaf(ar[k], we[k * C + c], s);
  float m = fmaxf(xs[(long long)src[e] * C + c] + s, 0.f) + 1e-7f;
  long long d = (long long)dst[e] * C + c;
  float ex = expf(m - mx[d]);
  atomicAdd(den + d, ex);
  atomicAdd(num + d, ex * m);
}

__global__ void node_update_kernel(const float* __restrict__ num,
                                   const float* __restrict__ den,
                                   const float* __restrict__ xd,
                                   float* __restrict__ h, int nc) {
  int t = blockIdx.x * blockDim.x + threadIdx.x;
  if (t >= nc) return;
  h[t] = num[t] / (den[t] + 1e-16f) + xd[t];
}

// ------------------------------- batchnorm ---------------------------------
__global__ void bn_stats_kernel(const float* __restrict__ h1, float* __restrict__ mu,
                                float* __restrict__ inv, int n, int C2) {
  int c = blockIdx.x * blockDim.x + threadIdx.x;
  if (c >= C2) return;
  float s = 0.f, q = 0.f;
  for (int i = 0; i < n; ++i) {
    float v = h1[(long long)i * C2 + c];
    s += v; q += v * v;
  }
  float m   = s / n;
  float var = q / n - m * m;
  mu[c]  = m;
  inv[c] = rsqrtf(var + 1e-5f);
}

__global__ void bn_apply_kernel(float* __restrict__ h1, const float* __restrict__ mu,
                                const float* __restrict__ inv, const float* __restrict__ g,
                                const float* __restrict__ bt, int total, int C2) {
  int t = blockIdx.x * blockDim.x + threadIdx.x;
  if (t >= total) return;
  int c = t % C2;
  h1[t] = fmaxf((h1[t] - mu[c]) * inv[c] * g[c] + bt[c], 0.f);
}

// ------------------------------- TopK pool ---------------------------------
__global__ void score_kernel(const float* __restrict__ x, const float* __restrict__ w,
                             float* __restrict__ score, int n, int C) {
  int i = blockIdx.x * blockDim.x + threadIdx.x;
  if (i >= n) return;
  float nrm = 0.f, s = 0.f;
  for (int c = 0; c < C; ++c) {
    nrm += w[c] * w[c];
    s   += x[(long long)i * C + c] * w[c];
  }
  score[i] = tanhf(s / sqrtf(nrm));
}

// one block per graph; iterative argmax in LDS; lower index wins ties (stable)
__global__ void topk_select_kernel(const float* __restrict__ score,
                                   float* __restrict__ vals, int* __restrict__ perm,
                                   int npg, int k) {
  __shared__ float sv[1024];
  __shared__ float rv[256];
  __shared__ int   ri[256];
  int g = blockIdx.x, t = threadIdx.x;
  for (int i = t; i < npg; i += blockDim.x) sv[i] = score[g * npg + i];
  __syncthreads();
  for (int j = 0; j < k; ++j) {
    float bv = -INFINITY; int bi = npg;
    for (int i = t; i < npg; i += blockDim.x) {
      float v = sv[i];
      if (v > bv || (v == bv && i < bi)) { bv = v; bi = i; }
    }
    rv[t] = bv; ri[t] = bi;
    __syncthreads();
    for (int st = 128; st > 0; st >>= 1) {
      if (t < st) {
        if (rv[t + st] > rv[t] || (rv[t + st] == rv[t] && ri[t + st] < ri[t])) {
          rv[t] = rv[t + st]; ri[t] = ri[t + st];
        }
      }
      __syncthreads();
    }
    if (t == 0) {
      int idx = ri[0];
      perm[g * k + j] = g * npg + idx;
      vals[g * k + j] = rv[0];
      sv[idx] = -INFINITY;
    }
    __syncthreads();
  }
}

__global__ void scatter_newpos_kernel(const int* __restrict__ perm,
                                      int* __restrict__ newpos, int nnew) {
  int i = blockIdx.x * blockDim.x + threadIdx.x;
  if (i < nnew) newpos[perm[i]] = i;
}

// newx = x[perm] * vals, then the forward's relu folded in
__global__ void gather_scale_relu_kernel(const float* __restrict__ xold,
                                         const int* __restrict__ perm,
                                         const float* __restrict__ vals,
                                         float* __restrict__ xnew, int total, int C) {
  int t = blockIdx.x * blockDim.x + threadIdx.x;
  if (t >= total) return;
  int i = t / C, c = t - i * C;
  float v = xold[(long long)perm[i] * C + c] * vals[i];
  xnew[t] = fmaxf(v, 0.f);
}

__global__ void remap_edges_kernel(const int* __restrict__ src, const int* __restrict__ dst,
                                   const int* __restrict__ ev, const int* __restrict__ newpos,
                                   int* __restrict__ nsrc, int* __restrict__ ndst,
                                   int* __restrict__ nev, int ne) {
  int e = blockIdx.x * blockDim.x + threadIdx.x;
  if (e >= ne) return;
  int ns = newpos[src[e]], nd = newpos[dst[e]];
  nev[e]  = (ev[e] && ns >= 0 && nd >= 0) ? 1 : 0;
  nsrc[e] = ns < 0 ? 0 : ns;
  ndst[e] = nd < 0 ? 0 : nd;
}

// ------------------------------ pool + head --------------------------------
__global__ void meanpool_kernel(const float* __restrict__ x, float* __restrict__ pooled,
                                int g_, int k, int C) {
  int t = blockIdx.x * blockDim.x + threadIdx.x;
  if (t >= g_ * C) return;
  int g = t / C, c = t - g * C;
  float s = 0.f;
  for (int i = 0; i < k; ++i) s += x[(long long)(g * k + i) * C + c];
  pooled[t] = s / (float)k;
}

__global__ void head_kernel(const float* __restrict__ h, const float* __restrict__ w,
                            const float* __restrict__ b, float* __restrict__ out,
                            int g_, int kdim) {
  int g = blockIdx.x * blockDim.x + threadIdx.x;
  if (g >= g_) return;
  float o[10];
  for (int j = 0; j < 10; ++j) {
    float s = b[j];
    for (int c = 0; c < kdim; ++c) s += h[g * kdim + c] * w[c * 10 + j];
    o[j] = s;
  }
  float m = o[0];
  for (int j = 1; j < 10; ++j) m = fmaxf(m, o[j]);
  float se = 0.f;
  for (int j = 0; j < 10; ++j) se += expf(o[j] - m);
  float lse = m + logf(se);
  for (int j = 0; j < 10; ++j) out[g * 10 + j] = o[j] - lse;
}

// --------------------------- host orchestration ----------------------------
static void run_conv(hipStream_t s, const float* x_in, int n, int cin, int C,
                     const float* ws, const float* bs, const float* wd, const float* bd,
                     const float* we, const float* be, const float* w1, const float* b1,
                     const float* g1, const float* bt1, const float* w2, const float* b2,
                     const int* src, const int* dst, const int* ev,
                     const float* eattr, int ne,
                     float* xs, float* xd, float* h, float* h1,
                     float* mx, float* den, float* num,
                     float* mu, float* inv, float* out) {
  const float* xs_u = x_in;
  const float* xd_u = x_in;
  if (ws) { launch_gemm(s, x_in, ws, bs, xs, n, C, cin); xs_u = xs; }
  if (wd) { launch_gemm(s, x_in, wd, bd, xd, n, C, cin); xd_u = xd; }
  int nc = n * C;
  fill_f32_kernel<<<cdiv(nc, 256), 256, 0, s>>>(mx, 0.f, nc);
  fill_f32_kernel<<<cdiv(nc, 256), 256, 0, s>>>(den, 0.f, nc);
  fill_f32_kernel<<<cdiv(nc, 256), 256, 0, s>>>(num, 0.f, nc);
  long long ec = (long long)ne * C;
  edge_max_kernel<<<cdiv(ec, 256), 256, 0, s>>>(xs_u, eattr, we, be, src, dst, ev, mx, ne, C);
  edge_sum_kernel<<<cdiv(ec, 256), 256, 0, s>>>(xs_u, eattr, we, be, src, dst, ev, mx, den, num, ne, C);
  node_update_kernel<<<cdiv(nc, 256), 256, 0, s>>>(num, den, xd_u, h, nc);
  launch_gemm(s, h, w1, b1, h1, n, 2 * C, C);
  bn_stats_kernel<<<cdiv(2 * C, 64), 64, 0, s>>>(h1, mu, inv, n, 2 * C);
  bn_apply_kernel<<<cdiv((long long)n * 2 * C, 256), 256, 0, s>>>(h1, mu, inv, g1, bt1, n * 2 * C, 2 * C);
  launch_gemm(s, h1, w2, b2, out, n, C, 2 * C);
}

static void run_topk(hipStream_t s, const float* x_in, int n_old, int g, int npg, int k, int C,
                     const float* wvec, const int* src, const int* dst, const int* ev,
                     int* nsrc, int* ndst, int* nev, int ne,
                     float* score, float* vals, int* perm, int* newpos, float* x_out) {
  score_kernel<<<cdiv(n_old, 256), 256, 0, s>>>(x_in, wvec, score, n_old, C);
  topk_select_kernel<<<g, 256, 0, s>>>(score, vals, perm, npg, k);
  fill_i32_kernel<<<cdiv(n_old, 256), 256, 0, s>>>(newpos, -1, n_old);
  int nnew = g * k;
  scatter_newpos_kernel<<<cdiv(nnew, 256), 256, 0, s>>>(perm, newpos, nnew);
  gather_scale_relu_kernel<<<cdiv((long long)nnew * C, 256), 256, 0, s>>>(x_in, perm, vals, x_out, nnew * C, C);
  remap_edges_kernel<<<cdiv(ne, 256), 256, 0, s>>>(src, dst, ev, newpos, nsrc, ndst, nev, ne);
}

extern "C" void kernel_launch(void* const* d_in, const int* in_sizes, int n_in,
                              void* d_out, int out_size, void* d_ws, size_t ws_size,
                              hipStream_t stream) {
  (void)in_sizes; (void)out_size; (void)ws_size;
  const float* x0 = (const float*)d_in[0];
  const int*   ei = (const int*)d_in[1];     // [2,E]: row0=src, row1=dst
  const float* ea = (const float*)d_in[2];   // [E,32]
  // params flattened in setup_inputs() dict insertion order:
  // conv1: ws bs wd bd we be w1 b1 g1 bt1 w2 b2   -> 4..15
  // conv2: we be w1 b1 g1 bt1 w2 b2               -> 16..23
  // conv3: ws bs wd bd we be w1 b1 g1 bt1 w2 b2   -> 24..35
  // p1_w 36, p2_w 37, p3_w 38, d1_w 39, d1_b 40, d2_w 41, d2_b 42
  const float* P[43] = {nullptr};
  for (int i = 4; i < 43 && i < n_in; ++i) P[i] = (const float*)d_in[i];

  // bump allocator over d_ws
  char* base = (char*)d_ws;
  size_t off = 0;
  auto A = [&](size_t bytes) -> void* {
    void* p = base + off;
    off += (bytes + 255) & ~(size_t)255;
    return p;
  };
  const size_t NC_MAX = (size_t)NN * 64;   // max n*C over all stages
  const size_t H1_MAX = (size_t)NN * 128;  // max n*2C

  float* xbufA  = (float*)A(NC_MAX * 4);
  float* xbufB  = (float*)A(NC_MAX * 4);
  float* xs     = (float*)A(NC_MAX * 4);
  float* xd     = (float*)A(NC_MAX * 4);
  float* hbuf   = (float*)A(NC_MAX * 4);
  float* h1     = (float*)A(H1_MAX * 4);
  float* mx     = (float*)A(NC_MAX * 4);
  float* den    = (float*)A(NC_MAX * 4);
  float* num    = (float*)A(NC_MAX * 4);
  int*   evA    = (int*)A((size_t)EE * 4);
  int*   srcB   = (int*)A((size_t)EE * 4);
  int*   dstB   = (int*)A((size_t)EE * 4);
  int*   evB    = (int*)A((size_t)EE * 4);
  int*   srcC   = (int*)A((size_t)EE * 4);
  int*   dstC   = (int*)A((size_t)EE * 4);
  int*   evC    = (int*)A((size_t)EE * 4);
  float* score  = (float*)A((size_t)NN * 4);
  float* vals   = (float*)A((size_t)NN * 4);
  int*   perm   = (int*)A((size_t)NN * 4);
  int*   newpos = (int*)A((size_t)NN * 4);
  float* mu     = (float*)A(256 * 4);
  float* inv    = (float*)A(256 * 4);
  float* pooled = (float*)A((size_t)GG * 128 * 4);
  float* dh     = (float*)A((size_t)GG * 64 * 4);

  const int* src0 = ei;
  const int* dst0 = ei + EE;
  fill_i32_kernel<<<cdiv(EE, 256), 256, 0, stream>>>(evA, 1, EE);

  // stage 1: GENConv(128 -> 64) on full graph, then TopK(0.2) -> 205/graph
  run_conv(stream, x0, NN, 128, 64,
           P[4], P[5], P[6], P[7], P[8], P[9], P[10], P[11], P[12], P[13], P[14], P[15],
           src0, dst0, evA, ea, EE,
           xs, xd, hbuf, h1, mx, den, num, mu, inv, xbufA);
  run_topk(stream, xbufA, NN, GG, NPG, KK1, 64, P[36], src0, dst0, evA,
           srcB, dstB, evB, EE, score, vals, perm, newpos, xbufB);

  // stage 2: GENConv(64 -> 64, no lin_src/lin_dst), TopK -> 41/graph
  run_conv(stream, xbufB, N1, 64, 64,
           nullptr, nullptr, nullptr, nullptr,
           P[16], P[17], P[18], P[19], P[20], P[21], P[22], P[23],
           srcB, dstB, evB, ea, EE,
           xs, xd, hbuf, h1, mx, den, num, mu, inv, xbufA);
  run_topk(stream, xbufA, N1, GG, KK1, KK2, 64, P[37], srcB, dstB, evB,
           srcC, dstC, evC, EE, score, vals, perm, newpos, xbufB);

  // stage 3: GENConv(64 -> 128), TopK -> 9/graph
  run_conv(stream, xbufB, N2, 64, 128,
           P[24], P[25], P[26], P[27], P[28], P[29], P[30], P[31], P[32], P[33], P[34], P[35],
           srcC, dstC, evC, ea, EE,
           xs, xd, hbuf, h1, mx, den, num, mu, inv, xbufA);
  run_topk(stream, xbufA, N2, GG, KK2, KK3, 128, P[38], srcC, dstC, evC,
           srcB, dstB, evB, EE, score, vals, perm, newpos, xbufB);

  // mean pool over K3=9 nodes/graph -> [128,128], dense head, log_softmax
  meanpool_kernel<<<cdiv(GG * 128, 256), 256, 0, stream>>>(xbufB, pooled, GG, KK3, 128);
  launch_gemm(stream, pooled, P[39], P[40], dh, GG, 64, 128);
  head_kernel<<<cdiv(GG, 64), 64, 0, stream>>>(dh, P[41], P[42], (float*)d_out, GG, 64);
}
